// FullAttenSUMNet_60730837566326
// MI455X (gfx1250) — matrix operand order repaired and checked by reading, not compiled, not run on hardware
//
#include <hip/hip_runtime.h>
#include <hip/hip_bf16.h>

// ---------------------------------------------------------------------------
// CDNA5 (gfx1250, wave32) WMMA types: v_wmma_f32_16x16x32_bf16
// ---------------------------------------------------------------------------
typedef __attribute__((ext_vector_type(16))) __bf16 bf16x16;
typedef __attribute__((ext_vector_type(8)))  float  floatx8;
typedef __attribute__((ext_vector_type(4)))  int    v4i;

#define AS_GLOBAL __attribute__((address_space(1)))
#define AS_LOCAL  __attribute__((address_space(3)))

// Async global->LDS path (ASYNCcnt-tracked), guarded so either toolchain compiles.
#if defined(__HIP_DEVICE_COMPILE__) && defined(__gfx1250__) && defined(__has_builtin)
#if __has_builtin(__builtin_amdgcn_global_load_async_to_lds_b128) && \
    __has_builtin(__builtin_amdgcn_s_wait_asynccnt)
#define USE_ASYNC_LDS 1
#endif
#endif

// LDS tile strides (elements). SA rows are 80B (16B aligned); SB rows 528B (16B aligned).
#define SA 40    // A tile: 64 rows x 32 K (bf16), padded
#define SB 132   // B tile: 32 K-rows x 128 N (f32), padded

// ---------------------------------------------------------------------------
// Blocked WMMA pointwise conv (1x1 conv == GEMM):
//   Y[b,co,h+pad,w+pad] = sum_ci W[co,ci] * X[b,ci,h,w]
// Block tile: M=64 x N=128, K-step 32, double-buffered LDS staging.
// 8 waves/block; wave (wm,wn) computes 16(M) x 64(N) via 4 accumulators.
//
// Operand packing used here (wave32; contiguous K, same order for A and B):
//   A (16x32 bf16): lane L -> M=L%16, g=L/16; elems e -> K=k0+g*16+e
//   B (32x16 bf16): lane L -> N=L%16, g=L/16; elems e -> K=k0+g*16+e
//   D (16x16 f32):  lane L -> N=L%16; VGPR r -> M=r+8*(L/16)
// ---------------------------------------------------------------------------
__device__ __forceinline__ void stage_tiles(
    const float* __restrict__ Wm, const float* __restrict__ X,
    __bf16* Al, float* Bl,
    int k0, int Ci, int Co, int m0, size_t xbase /* b0*Ci*HW + hw0 */,
    int HW, int tid)
{
    // ---- A tile: 64 x 32, f32 -> bf16, one 8-wide chunk per thread ----
    {
        const int m    = tid >> 2;
        const int koff = (tid & 3) * 8;
        const int mrow = m0 + m;
        const size_t wb = (size_t)mrow * Ci;
#pragma unroll
        for (int t = 0; t < 8; ++t) {
            const int k = k0 + koff + t;
            const float v = (mrow < Co && k < Ci) ? Wm[wb + k] : 0.0f;
            Al[m * SA + koff + t] = (__bf16)v;
        }
    }
    // ---- B tile: 32 K-rows x 128 N, raw f32, 4 x 16B chunks per thread ----
#pragma unroll
    for (int j = 0; j < 4; ++j) {
        const int c   = tid + j * 256;
        const int k   = c >> 5;           // 0..31
        const int col = (c & 31) * 4;     // 0..124
        float* dst = Bl + k * SB + col;
        if (k0 + k < Ci) {
            const float* src = X + xbase + (size_t)(k0 + k) * HW + col;
#ifdef USE_ASYNC_LDS
            __builtin_amdgcn_global_load_async_to_lds_b128(
                (AS_GLOBAL v4i*)(float*)src, (AS_LOCAL v4i*)dst, 0, 0);
#else
            const float4 v = *(const float4*)src;
            dst[0] = v.x; dst[1] = v.y; dst[2] = v.z; dst[3] = v.w;
#endif
        } else {
            dst[0] = 0.0f; dst[1] = 0.0f; dst[2] = 0.0f; dst[3] = 0.0f;
        }
    }
}

__global__ __launch_bounds__(256)
void wmma_pointwise_conv(const float* __restrict__ Wm, const float* __restrict__ X,
                         float* __restrict__ Y,
                         int Co, int Ci, int Bn, int H, int Wd, int pad)
{
    __shared__ __bf16 Al[2][64 * SA];
    __shared__ float  Bl[2][32 * SB];

    const int HW = H * Wd;
    const int N  = Bn * HW;
    const int tilesN = N >> 7;                 // N / 128
    const int m0 = (blockIdx.x / tilesN) * 64;
    const int n0 = (blockIdx.x % tilesN) * 128;

    const int tid  = threadIdx.x;
    const int w    = tid >> 5;
    const int wm   = w >> 1;                   // 0..3  (M subtile)
    const int wn   = w & 1;                    // 0..1  (N half)
    const int lane = tid & 31;
    const int g    = lane >> 4;
    const int lm   = lane & 15;

    // N-block always lies inside one batch image (HW % 128 == 0, HW >= 256)
    const int    b0    = n0 / HW;
    const int    hw0   = n0 % HW;
    const size_t xbase = (size_t)b0 * Ci * HW + hw0;

    const int nk = (Ci + 31) / 32;

    stage_tiles(Wm, X, Al[0], Bl[0], 0, Ci, Co, m0, xbase, HW, tid);

    const floatx8 zero = {};
    floatx8 acc[4];
#pragma unroll
    for (int s = 0; s < 4; ++s) acc[s] = zero;

    for (int i = 0; i < nk; ++i) {
        const int cur = i & 1;
#ifdef USE_ASYNC_LDS
        __builtin_amdgcn_s_wait_asynccnt(0);
#endif
        __syncthreads();
        if (i + 1 < nk)
            stage_tiles(Wm, X, Al[cur ^ 1], Bl[cur ^ 1],
                        (i + 1) * 32, Ci, Co, m0, xbase, HW, tid);

        // A fragment (shared across the wave's 4 N-subtiles)
        bf16x16 a;
        const __bf16* arow = &Al[cur][(wm * 16 + lm) * SA + g * 16];
#pragma unroll
        for (int t = 0; t < 8; ++t) {
            a[t]     = arow[t];
            a[t + 8] = arow[8 + t];
        }
        // 4 x (B fragment + WMMA)
#pragma unroll
        for (int s = 0; s < 4; ++s) {
            bf16x16 bm;
            const float* bcol = &Bl[cur][(g * 16) * SB + wn * 64 + s * 16 + lm];
#pragma unroll
            for (int e = 0; e < 16; ++e)
                bm[e] = (__bf16)bcol[e * SB];
            acc[s] = __builtin_amdgcn_wmma_f32_16x16x32_bf16(
                false, a, false, bm, (short)0, acc[s], false, false);
        }
    }

    // Store D into (optionally padded) output layout
    const int Hp = H + 2 * pad, Wp = Wd + 2 * pad;
#pragma unroll
    for (int s = 0; s < 4; ++s) {
        const int n_g = n0 + wn * 64 + s * 16 + lm;
        const int nhw = n_g % HW;
        const int nb  = n_g / HW;
        const int oh  = nhw / Wd + pad;
        const int ow  = nhw % Wd + pad;
#pragma unroll
        for (int r = 0; r < 8; ++r) {
            const int mo = m0 + wm * 16 + r + 8 * g;
            if (mo < Co)
                Y[(((size_t)nb * Co + mo) * Hp + oh) * Wp + ow] = acc[s][r];
        }
    }
}

// ---------------------------------------------------------------------------
// Fused 3x3-window attention: softmax(q * (k+relbias)) . v   (+ optional relu)
// q: [B,Co,H,W]; kp/vp: [B,Co,H+2,W+2] zero-padded; rh/rw: [Co/2, 3]
// ---------------------------------------------------------------------------
__global__ __launch_bounds__(256)
void attn_combine(const float* __restrict__ q, const float* __restrict__ kp,
                  const float* __restrict__ vp, const float* __restrict__ rh,
                  const float* __restrict__ rw, float* __restrict__ out,
                  int Co, int Bn, int H, int Wd, int do_relu)
{
    const int i = blockIdx.x * blockDim.x + threadIdx.x;
    const int total = Bn * Co * H * Wd;
    if (i >= total) return;
    const int w = i % Wd;
    const int h = (i / Wd) % H;
    const int c = (i / (H * Wd)) % Co;
    const int b = i / (Co * H * Wd);
    const int Hp = H + 2, Wp = Wd + 2, half = Co >> 1;

    const float qv = q[i];
    const size_t base = (((size_t)b * Co + c) * Hp + h) * Wp + w;

    float lg[9], vv[9];
    float mx = -3.0e38f;
#pragma unroll
    for (int dy = 0; dy < 3; ++dy) {
#pragma unroll
        for (int dx = 0; dx < 3; ++dx) {
            const int t = dy * 3 + dx;
            float kk = kp[base + (size_t)dy * Wp + dx];
            kk += (c < half) ? rh[c * 3 + dy] : rw[(c - half) * 3 + dx];
            const float l = qv * kk;
            lg[t] = l;
            vv[t] = vp[base + (size_t)dy * Wp + dx];
            mx = fmaxf(mx, l);
        }
    }
    float se = 0.0f, acc = 0.0f;
#pragma unroll
    for (int t = 0; t < 9; ++t) {
        const float e = __expf(lg[t] - mx);
        se  += e;
        acc += e * vv[t];
    }
    float o = acc / se;
    if (do_relu) o = fmaxf(o, 0.0f);
    out[i] = o;
}

// ---------------------------------------------------------------------------
// BatchNorm (train-mode stats over B,H,W) + ReLU. One block per channel.
// ---------------------------------------------------------------------------
__global__ __launch_bounds__(256)
void bn_relu(const float* __restrict__ x, float* __restrict__ y,
             const float* __restrict__ gamma, const float* __restrict__ beta,
             int C, int Bn, int HW)
{
    const int c = blockIdx.x;
    const int n = Bn * HW;
    __shared__ float ssum[256];
    __shared__ float ssq[256];
    float s = 0.0f, q = 0.0f;
    for (int t = threadIdx.x; t < n; t += blockDim.x) {
        const int b = t / HW, j = t % HW;
        const float v = x[((size_t)b * C + c) * HW + j];
        s += v; q += v * v;
    }
    ssum[threadIdx.x] = s; ssq[threadIdx.x] = q;
    __syncthreads();
    for (int off = 128; off > 0; off >>= 1) {
        if ((int)threadIdx.x < off) {
            ssum[threadIdx.x] += ssum[threadIdx.x + off];
            ssq[threadIdx.x]  += ssq[threadIdx.x + off];
        }
        __syncthreads();
    }
    const float mean = ssum[0] / (float)n;
    const float var  = ssq[0] / (float)n - mean * mean;
    const float sc = gamma[c] * rsqrtf(var + 1e-5f);
    const float sh = beta[c] - mean * sc;
    for (int t = threadIdx.x; t < n; t += blockDim.x) {
        const int b = t / HW, j = t % HW;
        const size_t a = ((size_t)b * C + c) * HW + j;
        y[a] = fmaxf(x[a] * sc + sh, 0.0f);
    }
}

// ---------------------------------------------------------------------------
// 2x2 maxpool with first-max argmax (idx = row*2+col within window)
// ---------------------------------------------------------------------------
__global__ __launch_bounds__(256)
void maxpool2k(const float* __restrict__ x, float* __restrict__ y,
               int* __restrict__ idx, int C, int Bn, int H, int Wd)
{
    const int H2 = H >> 1, W2 = Wd >> 1;
    const int i = blockIdx.x * blockDim.x + threadIdx.x;
    const int total = Bn * C * H2 * W2;
    if (i >= total) return;
    const int w2 = i % W2;
    const int h2 = (i / W2) % H2;
    const int c  = (i / (H2 * W2)) % C;
    const int b  = i / (C * H2 * W2);
    const size_t base = (((size_t)b * C + c) * H + 2 * h2) * Wd + 2 * w2;
    float best = x[base]; int bi = 0;
    float v = x[base + 1];      if (v > best) { best = v; bi = 1; }
    v = x[base + Wd];           if (v > best) { best = v; bi = 2; }
    v = x[base + Wd + 1];       if (v > best) { best = v; bi = 3; }
    y[i] = best; idx[i] = bi;
}

// ---------------------------------------------------------------------------
// Unpool into channels [c0, c0+C) of a concat buffer with Ctot channels
// ---------------------------------------------------------------------------
__global__ __launch_bounds__(256)
void unpool2k(const float* __restrict__ v, const int* __restrict__ idx,
              float* __restrict__ dst, int C, int Ctot, int c0,
              int Bn, int H2, int W2)
{
    const int i = blockIdx.x * blockDim.x + threadIdx.x;
    const int total = Bn * C * H2 * W2;
    if (i >= total) return;
    const int w2 = i % W2;
    const int h2 = (i / W2) % H2;
    const int c  = (i / (H2 * W2)) % C;
    const int b  = i / (C * H2 * W2);
    const int H = H2 * 2, Wd = W2 * 2;
    const float val = v[i];
    const int id = idx[i];
    const size_t base = (((size_t)b * Ctot + c0 + c) * H + 2 * h2) * Wd + 2 * w2;
    dst[base]          = (id == 0) ? val : 0.0f;
    dst[base + 1]      = (id == 1) ? val : 0.0f;
    dst[base + Wd]     = (id == 2) ? val : 0.0f;
    dst[base + Wd + 1] = (id == 3) ? val : 0.0f;
}

// Copy src [B,C,HW] into channels [c0, c0+C) of dst with Ctot channels
__global__ __launch_bounds__(256)
void copy_cat(const float* __restrict__ src, float* __restrict__ dst,
              int C, int Ctot, int c0, int Bn, int HW)
{
    const int i = blockIdx.x * blockDim.x + threadIdx.x;
    const int total = Bn * C * HW;
    if (i >= total) return;
    const int j = i % HW;
    const int c = (i / HW) % C;
    const int b = i / (C * HW);
    dst[((size_t)b * Ctot + c0 + c) * HW + j] = src[i];
}

// Output layer: ks=1 attention degenerates to out = Wv . x  (Co=2, Ci=32)
__global__ __launch_bounds__(256)
void out_conv(const float* __restrict__ Wv, const float* __restrict__ x,
              float* __restrict__ out, int Ci, int Co, int Bn, int HW)
{
    const int i = blockIdx.x * blockDim.x + threadIdx.x;
    const int total = Bn * Co * HW;
    if (i >= total) return;
    const int j = i % HW;
    const int o = (i / HW) % Co;
    const int b = i / (Co * HW);
    float s = 0.0f;
    for (int ci = 0; ci < Ci; ++ci)
        s += Wv[o * Ci + ci] * x[((size_t)b * Ci + ci) * HW + j];
    out[i] = s;
}

// ---------------------------------------------------------------------------
// Host-side helpers
// ---------------------------------------------------------------------------
static inline void run_attn(const float* wq, const float* wk, const float* wv,
                            const float* rh, const float* rw,
                            const float* X, float* Y,
                            int Ci, int Co, int Bn, int H, int Wd, int relu,
                            float* sq, float* sk, float* sv, hipStream_t stream)
{
    const int HW = H * Wd, N = Bn * HW, Hp = H + 2, Wp = Wd + 2;
    (void)hipMemsetAsync(sk, 0, (size_t)Bn * Co * Hp * Wp * sizeof(float), stream);
    (void)hipMemsetAsync(sv, 0, (size_t)Bn * Co * Hp * Wp * sizeof(float), stream);
    const int blocks = ((Co + 63) / 64) * (N / 128);
    wmma_pointwise_conv<<<blocks, 256, 0, stream>>>(wq, X, sq, Co, Ci, Bn, H, Wd, 0);
    wmma_pointwise_conv<<<blocks, 256, 0, stream>>>(wk, X, sk, Co, Ci, Bn, H, Wd, 1);
    wmma_pointwise_conv<<<blocks, 256, 0, stream>>>(wv, X, sv, Co, Ci, Bn, H, Wd, 1);
    const int tot = Bn * Co * HW;
    attn_combine<<<(tot + 255) / 256, 256, 0, stream>>>(sq, sk, sv, rh, rw, Y,
                                                        Co, Bn, H, Wd, relu);
}

extern "C" void kernel_launch(void* const* d_in, const int* in_sizes, int n_in,
                              void* d_out, int out_size, void* d_ws, size_t ws_size,
                              hipStream_t stream)
{
    (void)in_sizes; (void)n_in; (void)out_size; (void)ws_size;
    const int Bn = 2;

    const float* x = (const float*)d_in[0];
    // 17 attention layers x {wq, wk, wv, rel_h, rel_w}
    const float* P[17][5];
    for (int i = 0; i < 17; ++i)
        for (int j = 0; j < 5; ++j)
            P[i][j] = (const float*)d_in[1 + i * 5 + j];
    // 8 BN layers x {g, b}
    const float* Gm[8]; const float* Bt[8];
    for (int j = 0; j < 8; ++j) {
        Gm[j] = (const float*)d_in[86 + 2 * j];
        Bt[j] = (const float*)d_in[87 + 2 * j];
    }

    // Workspace bump allocator (256B aligned)
    size_t off = 0;
    auto allocF = [&](size_t elems) -> float* {
        float* p = (float*)((char*)d_ws + off);
        off += ((elems * sizeof(float) + 255) & ~(size_t)255);
        return p;
    };
    auto allocI = [&](size_t elems) -> int* {
        int* p = (int*)((char*)d_ws + off);
        off += ((elems * sizeof(int) + 255) & ~(size_t)255);
        return p;
    };

    // attention scratch (sized for the largest layer: conv2/donv2 @128x128)
    float* S_Q = allocF(4194304);
    float* S_K = allocF(4326400);   // 128*2*130*130
    float* S_V = allocF(4326400);
    float* S_A = allocF(4194304);   // pre-BN attention output

    float* conv1  = allocF(2097152);
    float* conv2  = allocF(4194304);
    float* pool1  = allocF(1048576);  int* idx1 = allocI(1048576);
    float* conv3a = allocF(2097152);
    float* conv3b = allocF(2097152);
    float* pool2  = allocF(524288);   int* idx2 = allocI(524288);
    float* conv4a = allocF(1048576);
    float* conv4b = allocF(1048576);
    float* pool3  = allocF(262144);   int* idx3 = allocI(262144);
    float* conv5a = allocF(262144);
    float* conv5b = allocF(262144);
    float* pool4  = allocF(65536);    int* idx4 = allocI(65536);
    float* up4    = allocF(524288);
    float* d5b    = allocF(262144);
    float* d5a    = allocF(262144);
    float* up3    = allocF(2097152);
    float* d4b    = allocF(1048576);
    float* d4a    = allocF(524288);
    float* up2    = allocF(4194304);
    float* d3b    = allocF(2097152);
    float* d3a    = allocF(1048576);
    float* up1    = allocF(8388608);
    float* d2     = allocF(2097152);
    float* cat1   = allocF(4194304);
    float* d1     = allocF(1048576);

    auto grid = [](int tot) { return (tot + 255) / 256; };

    // ------------------------- encoder -------------------------
    run_attn(P[0][0], P[0][1], P[0][2], P[0][3], P[0][4], x, S_A,
             1, 64, Bn, 128, 128, 0, S_Q, S_K, S_V, stream);
    bn_relu<<<64, 256, 0, stream>>>(S_A, conv1, Gm[0], Bt[0], 64, Bn, 16384);
    run_attn(P[1][0], P[1][1], P[1][2], P[1][3], P[1][4], conv1, S_A,
             64, 128, Bn, 128, 128, 0, S_Q, S_K, S_V, stream);
    bn_relu<<<128, 256, 0, stream>>>(S_A, conv2, Gm[1], Bt[1], 128, Bn, 16384);
    maxpool2k<<<grid(Bn * 128 * 64 * 64), 256, 0, stream>>>(conv2, pool1, idx1, 128, Bn, 128, 128);
    run_attn(P[2][0], P[2][1], P[2][2], P[2][3], P[2][4], pool1, S_A,
             128, 256, Bn, 64, 64, 0, S_Q, S_K, S_V, stream);
    bn_relu<<<256, 256, 0, stream>>>(S_A, conv3a, Gm[2], Bt[2], 256, Bn, 4096);
    run_attn(P[3][0], P[3][1], P[3][2], P[3][3], P[3][4], conv3a, S_A,
             256, 256, Bn, 64, 64, 0, S_Q, S_K, S_V, stream);
    bn_relu<<<256, 256, 0, stream>>>(S_A, conv3b, Gm[3], Bt[3], 256, Bn, 4096);
    maxpool2k<<<grid(Bn * 256 * 32 * 32), 256, 0, stream>>>(conv3b, pool2, idx2, 256, Bn, 64, 64);
    run_attn(P[4][0], P[4][1], P[4][2], P[4][3], P[4][4], pool2, S_A,
             256, 512, Bn, 32, 32, 0, S_Q, S_K, S_V, stream);
    bn_relu<<<512, 256, 0, stream>>>(S_A, conv4a, Gm[4], Bt[4], 512, Bn, 1024);
    run_attn(P[5][0], P[5][1], P[5][2], P[5][3], P[5][4], conv4a, S_A,
             512, 512, Bn, 32, 32, 0, S_Q, S_K, S_V, stream);
    bn_relu<<<512, 256, 0, stream>>>(S_A, conv4b, Gm[5], Bt[5], 512, Bn, 1024);
    maxpool2k<<<grid(Bn * 512 * 16 * 16), 256, 0, stream>>>(conv4b, pool3, idx3, 512, Bn, 32, 32);
    run_attn(P[6][0], P[6][1], P[6][2], P[6][3], P[6][4], pool3, S_A,
             512, 512, Bn, 16, 16, 0, S_Q, S_K, S_V, stream);
    bn_relu<<<512, 256, 0, stream>>>(S_A, conv5a, Gm[6], Bt[6], 512, Bn, 256);
    run_attn(P[7][0], P[7][1], P[7][2], P[7][3], P[7][4], conv5a, S_A,
             512, 512, Bn, 16, 16, 0, S_Q, S_K, S_V, stream);
    bn_relu<<<512, 256, 0, stream>>>(S_A, conv5b, Gm[7], Bt[7], 512, Bn, 256);
    maxpool2k<<<grid(Bn * 512 * 8 * 8), 256, 0, stream>>>(conv5b, pool4, idx4, 512, Bn, 16, 16);

    // ------------------------- decoder -------------------------
    unpool2k<<<grid(Bn * 512 * 8 * 8), 256, 0, stream>>>(pool4, idx4, up4, 512, 1024, 0, Bn, 8, 8);
    copy_cat<<<grid(Bn * 512 * 256), 256, 0, stream>>>(conv5b, up4, 512, 1024, 512, Bn, 256);
    run_attn(P[8][0], P[8][1], P[8][2], P[8][3], P[8][4], up4, d5b,
             1024, 512, Bn, 16, 16, 1, S_Q, S_K, S_V, stream);
    run_attn(P[9][0], P[9][1], P[9][2], P[9][3], P[9][4], d5b, d5a,
             512, 512, Bn, 16, 16, 1, S_Q, S_K, S_V, stream);
    unpool2k<<<grid(Bn * 512 * 16 * 16), 256, 0, stream>>>(d5a, idx3, up3, 512, 1024, 0, Bn, 16, 16);
    copy_cat<<<grid(Bn * 512 * 1024), 256, 0, stream>>>(conv4b, up3, 512, 1024, 512, Bn, 1024);
    run_attn(P[10][0], P[10][1], P[10][2], P[10][3], P[10][4], up3, d4b,
             1024, 512, Bn, 32, 32, 1, S_Q, S_K, S_V, stream);
    run_attn(P[11][0], P[11][1], P[11][2], P[11][3], P[11][4], d4b, d4a,
             512, 256, Bn, 32, 32, 1, S_Q, S_K, S_V, stream);
    unpool2k<<<grid(Bn * 256 * 32 * 32), 256, 0, stream>>>(d4a, idx2, up2, 256, 512, 0, Bn, 32, 32);
    copy_cat<<<grid(Bn * 256 * 4096), 256, 0, stream>>>(conv3b, up2, 256, 512, 256, Bn, 4096);
    run_attn(P[12][0], P[12][1], P[12][2], P[12][3], P[12][4], up2, d3b,
             512, 256, Bn, 64, 64, 1, S_Q, S_K, S_V, stream);
    run_attn(P[13][0], P[13][1], P[13][2], P[13][3], P[13][4], d3b, d3a,
             256, 128, Bn, 64, 64, 1, S_Q, S_K, S_V, stream);
    unpool2k<<<grid(Bn * 128 * 64 * 64), 256, 0, stream>>>(d3a, idx1, up1, 128, 256, 0, Bn, 64, 64);
    copy_cat<<<grid(Bn * 128 * 16384), 256, 0, stream>>>(conv2, up1, 128, 256, 128, Bn, 16384);
    run_attn(P[14][0], P[14][1], P[14][2], P[14][3], P[14][4], up1, d2,
             256, 64, Bn, 128, 128, 1, S_Q, S_K, S_V, stream);
    copy_cat<<<grid(Bn * 64 * 16384), 256, 0, stream>>>(d2, cat1, 64, 128, 0, Bn, 16384);
    copy_cat<<<grid(Bn * 64 * 16384), 256, 0, stream>>>(conv1, cat1, 64, 128, 64, Bn, 16384);
    run_attn(P[15][0], P[15][1], P[15][2], P[15][3], P[15][4], cat1, d1,
             128, 32, Bn, 128, 128, 1, S_Q, S_K, S_V, stream);

    // output layer (ks=1, softmax over singleton => out = Wv . d1)
    out_conv<<<grid(Bn * 2 * 16384), 256, 0, stream>>>(P[16][2], d1, (float*)d_out,
                                                       32, 2, Bn, 16384);
}
